// EGNNBlock_84885733638847
// MI455X (gfx1250) — matrix operand order repaired and checked by reading, not compiled, not run on hardware
//
#include <hip/hip_runtime.h>
#include <hip/hip_bf16.h>

// ---------------------------------------------------------------------------
// EGNN block for MI455X (gfx1250, wave32, WMMA).
// Dominant cost: edge-message GEMM [320000 x 336] @ [336 x 256] x 4 layers.
// Strategy: bf16 WMMA (16x16x32, f32 accum) with fused atomic scatter-sum.
// ---------------------------------------------------------------------------

typedef __attribute__((ext_vector_type(16))) __bf16 v16bf;
typedef __attribute__((ext_vector_type(8)))  __bf16 v8bf;
typedef __attribute__((ext_vector_type(8)))  float  v8f;

#define N_STEP   64
#define GAMMA    10.0f
#define N_LAYER  4
#define H_DIM    128
#define E_DIM    16
#define HID_DIM  256
#define IN_M     336          // 2*H_DIM + E_DIM + N_STEP
#define KC_TILES 11           // ceil(336/32)
#define K_PAD    (KC_TILES*32) // 352
#define N_NODES  10000
#define N_EDGES  320000
#define SLOPE    0.01f

__device__ __forceinline__ void fatomic_add(float* p, float v) {
    __hip_atomic_fetch_add(p, v, __ATOMIC_RELAXED, __HIP_MEMORY_SCOPE_AGENT);
}
__device__ __forceinline__ float lrelu(float v) { return v > 0.f ? v : SLOPE * v; }

// ---------------------------------------------------------------------------
// Pack Wm (f32, [L, 336, 256]) into bf16 in exact WMMA-B lane order:
// packed[((layer*11 + kc)*16 + ntile)*32 + lane][h], h=0..15,
//   K = kc*32 + (h<8 ? 0:16) + (lane>=16 ? 8:0) + (h&7),  N = ntile*16 + (lane&15),
// zero-padded for K >= 336.  Each lane then loads 32 contiguous bytes.
// ---------------------------------------------------------------------------
__global__ void pack_wm_kernel(const float* __restrict__ Wm, __bf16* __restrict__ packed) {
    const int total = N_LAYER * KC_TILES * 16 * 32 * 16;
    int t = blockIdx.x * blockDim.x + threadIdx.x;
    if (t >= total) return;
    int h = t & 15;
    int r = t >> 4;
    int lane = r & 31; r >>= 5;
    int nt = r & 15;   r >>= 4;
    int kc = r % KC_TILES;
    int l  = r / KC_TILES;
    int k = kc * 32 + ((h >= 8) ? 16 : 0) + ((lane >= 16) ? 8 : 0) + (h & 7);
    int n = nt * 16 + (lane & 15);
    float v = (k < IN_M) ? Wm[((size_t)l * IN_M + k) * HID_DIM + n] : 0.f;
    packed[t] = (__bf16)v;
}

// Degree of each destination node (layer-invariant).
__global__ void count_kernel(const int* __restrict__ ei, float* __restrict__ cnt) {
    int e = blockIdx.x * blockDim.x + threadIdx.x;
    if (e < N_EDGES) fatomic_add(&cnt[ei[e]], 1.0f);
}

// ---------------------------------------------------------------------------
// Edge kernel: one block = 16 edges. Phase 1 builds the 16x352 bf16 feature
// tile in LDS; phase 2 is the WMMA GEMM (8 waves x 2 N-tiles = 256 cols)
// with fused bias + atomic scatter-sum into m_aggr. Also scatters pos_vect.
// ---------------------------------------------------------------------------
__global__ void __launch_bounds__(256)
edge_kernel(const float* __restrict__ x, const float* __restrict__ pos,
            const int* __restrict__ ei, const float* __restrict__ ea,
            const __bf16* __restrict__ wmP,   // this layer: [11][16][32][16] bf16
            const float* __restrict__ bm,     // this layer: [256]
            float* __restrict__ m_aggr,       // [N_NODES, 256]
            float* __restrict__ pos_aggr)     // [N_NODES, 3]
{
    __shared__ __bf16 feat[16][K_PAD];
    __shared__ int sI[16];

    const int tid = threadIdx.x;
    const int g = tid >> 4;       // edge slot 0..15
    const int t = tid & 15;       // lane within 16-lane group
    const int e = blockIdx.x * 16 + g;

    const int i = ei[e];
    const int j = ei[N_EDGES + e];
    if (t == 0) sI[g] = i;

    // ---- Phase 1: features -------------------------------------------------
    const float4* xi4 = reinterpret_cast<const float4*>(x + (size_t)i * H_DIM) + t * 2;
    const float4* xj4 = reinterpret_cast<const float4*>(x + (size_t)j * H_DIM) + t * 2;
    float s = 0.f;
    #pragma unroll
    for (int q = 0; q < 2; ++q) {
        float4 a = xi4[q];
        float4 b = xj4[q];
        int c = t * 8 + q * 4;
        feat[g][c + 0] = (__bf16)a.x; feat[g][c + 1] = (__bf16)a.y;
        feat[g][c + 2] = (__bf16)a.z; feat[g][c + 3] = (__bf16)a.w;
        feat[g][H_DIM + c + 0] = (__bf16)b.x; feat[g][H_DIM + c + 1] = (__bf16)b.y;
        feat[g][H_DIM + c + 2] = (__bf16)b.z; feat[g][H_DIM + c + 3] = (__bf16)b.w;
        float dx = a.x - b.x, dy = a.y - b.y, dz = a.z - b.z, dw = a.w - b.w;
        s += dx * dx + dy * dy + dz * dz + dw * dw;
    }
    #pragma unroll
    for (int m = 8; m >= 1; m >>= 1) s += __shfl_xor(s, m, 16);
    float rdist = sqrtf(s);

    #pragma unroll
    for (int q = 0; q < 4; ++q) {   // 64 RBF features, linspace(0,10,64)
        int kk = t * 4 + q;
        float d = rdist - (10.0f / 63.0f) * (float)kk;
        feat[g][2 * H_DIM + kk] = (__bf16)__expf(-GAMMA * d * d);
    }
    feat[g][2 * H_DIM + N_STEP + t] = (__bf16)ea[(size_t)e * E_DIM + t];
    feat[g][IN_M + t] = (__bf16)0.f;  // pad K 336..351

    if (t < 3) {  // pos_vect scatter (mean numerator)
        float pv = pos[(size_t)i * 3 + t] - pos[(size_t)j * 3 + t];
        fatomic_add(&pos_aggr[(size_t)i * 3 + t], pv);
    }
    __syncthreads();

    // ---- Phase 2: WMMA GEMM (no divergence; EXEC all ones) -----------------
    const int wv   = tid >> 5;     // wave 0..7 -> N-tiles wv and wv+8
    const int lane = tid & 31;
    const int m    = lane & 15;
    const int hi   = lane >> 4;    // 0 | 1
    const int kOff = hi * 8;       // A layout: lanes0-15 K{0..7,16..23}, lanes16-31 K{8..15,24..31}

    v8f acc0 = {};
    v8f acc1 = {};
    for (int kc = 0; kc < KC_TILES; ++kc) {
        union { v16bf v; v8bf h[2]; } a;
        const __bf16* arow = &feat[m][kc * 32 + kOff];
        a.h[0] = *reinterpret_cast<const v8bf*>(arow);        // ds_load_b128
        a.h[1] = *reinterpret_cast<const v8bf*>(arow + 16);   // ds_load_b128

        const __bf16* b0p = wmP + (((size_t)kc * 16 + wv)     * 32 + lane) * 16;
        const __bf16* b1p = wmP + (((size_t)kc * 16 + wv + 8) * 32 + lane) * 16;
        if (kc + 1 < KC_TILES)
            __builtin_prefetch(wmP + (((size_t)(kc + 1) * 16 + wv) * 32 + lane) * 16, 0, 3);
        v16bf b0 = *reinterpret_cast<const v16bf*>(b0p);
        v16bf b1 = *reinterpret_cast<const v16bf*>(b1p);

        acc0 = __builtin_amdgcn_wmma_f32_16x16x32_bf16(false, a.v, false, b0,
                                                       (short)0, acc0, false, false);
        acc1 = __builtin_amdgcn_wmma_f32_16x16x32_bf16(false, a.v, false, b1,
                                                       (short)0, acc1, false, false);
    }

    // C/D layout: VGPR r, lanes0-15 -> M=r,N=lane ; lanes16-31 -> M=r+8,N=lane-16
    const int ncol0 = wv * 16 + (lane & 15);
    const int ncol1 = (wv + 8) * 16 + (lane & 15);
    const float bm0 = bm[ncol0];
    const float bm1 = bm[ncol1];
    #pragma unroll
    for (int r0 = 0; r0 < 8; ++r0) {
        int row = r0 + 8 * hi;
        int dst = sI[row];
        fatomic_add(&m_aggr[(size_t)dst * HID_DIM + ncol0], acc0[r0] + bm0);
        fatomic_add(&m_aggr[(size_t)dst * HID_DIM + ncol1], acc1[r0] + bm1);
    }
}

// ---------------------------------------------------------------------------
// Node update: pos += pos_aggr/max(cnt,1); x = lrelu([x, m_aggr] @ Wx + bx).
// One 128-thread block per node; Wx reads are coalesced and L2-resident.
// ---------------------------------------------------------------------------
__global__ void __launch_bounds__(128)
node_kernel(const float* __restrict__ x, const float* __restrict__ pos,
            const float* __restrict__ m_aggr, const float* __restrict__ pos_aggr,
            const float* __restrict__ cnt,
            const float* __restrict__ Wx,  // this layer: [384, 128]
            const float* __restrict__ bx,  // this layer: [128]
            float* __restrict__ x_out, float* __restrict__ pos_out)
{
    __shared__ float feat[H_DIM + HID_DIM];
    const int nd = blockIdx.x;
    const int c  = threadIdx.x;
    feat[c]               = x[(size_t)nd * H_DIM + c];
    feat[H_DIM + c]       = m_aggr[(size_t)nd * HID_DIM + c];
    feat[2 * H_DIM + c]   = m_aggr[(size_t)nd * HID_DIM + H_DIM + c];
    __syncthreads();

    float acc = bx[c];
    #pragma unroll 8
    for (int k = 0; k < H_DIM + HID_DIM; ++k)
        acc = fmaf(feat[k], Wx[(size_t)k * H_DIM + c], acc);
    x_out[(size_t)nd * H_DIM + c] = lrelu(acc);

    if (c < 3) {
        float cn = fmaxf(cnt[nd], 1.0f);
        pos_out[(size_t)nd * 3 + c] =
            pos[(size_t)nd * 3 + c] + pos_aggr[(size_t)nd * 3 + c] / cn;
    }
}

// Residual add + final leaky_relu; emit (x, pos) concatenated.
__global__ void final_kernel(const float* __restrict__ x_ori, const float* __restrict__ x,
                             const float* __restrict__ pos, float* __restrict__ out) {
    int t = blockIdx.x * blockDim.x + threadIdx.x;
    if (t < N_NODES * H_DIM) out[t] = lrelu(x_ori[t] + x[t]);
    if (t < N_NODES * 3)     out[N_NODES * H_DIM + t] = pos[t];
}

// ---------------------------------------------------------------------------
extern "C" void kernel_launch(void* const* d_in, const int* in_sizes, int n_in,
                              void* d_out, int out_size, void* d_ws, size_t ws_size,
                              hipStream_t stream) {
    (void)in_sizes; (void)n_in; (void)out_size; (void)ws_size;
    const float* x_in   = (const float*)d_in[0];
    const float* pos_in = (const float*)d_in[1];
    const int*   ei     = (const int*)  d_in[2];
    const float* ea     = (const float*)d_in[3];
    const float* Wm     = (const float*)d_in[4];
    const float* bm     = (const float*)d_in[5];
    // d_in[6] (Wp), d_in[7] (bp): dead in the reference -- skipped.
    const float* Wx     = (const float*)d_in[8];
    const float* bx     = (const float*)d_in[9];
    float* out = (float*)d_out;

    char* ws = (char*)d_ws;
    size_t off = 0;
    auto carve = [&](size_t bytes) -> void* {
        void* p = (void*)(ws + off);
        off += (bytes + 255) & ~(size_t)255;
        return p;
    };
    float* xbuf0    = (float*)carve((size_t)N_NODES * H_DIM * sizeof(float));
    float* xbuf1    = (float*)carve((size_t)N_NODES * H_DIM * sizeof(float));
    float* posbuf0  = (float*)carve((size_t)N_NODES * 3 * sizeof(float));
    float* posbuf1  = (float*)carve((size_t)N_NODES * 3 * sizeof(float));
    float* m_aggr   = (float*)carve((size_t)N_NODES * HID_DIM * sizeof(float));
    float* pos_aggr = (float*)carve((size_t)N_NODES * 3 * sizeof(float));
    float* cnt      = (float*)carve((size_t)N_NODES * sizeof(float));
    __bf16* wmP     = (__bf16*)carve((size_t)N_LAYER * KC_TILES * 16 * 32 * 16 * sizeof(__bf16));
    float* xbufs[2]   = {xbuf0, xbuf1};
    float* posbufs[2] = {posbuf0, posbuf1};

    {   // One-shot weight pack (deterministic; cheap: ~360K elements)
        const int total = N_LAYER * KC_TILES * 16 * 32 * 16;
        pack_wm_kernel<<<(total + 255) / 256, 256, 0, stream>>>(Wm, wmP);
    }
    hipMemsetAsync(cnt, 0, (size_t)N_NODES * sizeof(float), stream);
    count_kernel<<<(N_EDGES + 255) / 256, 256, 0, stream>>>(ei, cnt);

    const float* xc = x_in;
    const float* pc = pos_in;
    for (int l = 0; l < N_LAYER; ++l) {
        hipMemsetAsync(m_aggr, 0, (size_t)N_NODES * HID_DIM * sizeof(float), stream);
        hipMemsetAsync(pos_aggr, 0, (size_t)N_NODES * 3 * sizeof(float), stream);
        edge_kernel<<<N_EDGES / 16, 256, 0, stream>>>(
            xc, pc, ei, ea,
            wmP + (size_t)l * KC_TILES * 16 * 32 * 16,
            bm + (size_t)l * HID_DIM,
            m_aggr, pos_aggr);
        node_kernel<<<N_NODES, 128, 0, stream>>>(
            xc, pc, m_aggr, pos_aggr, cnt,
            Wx + (size_t)l * (H_DIM + HID_DIM) * H_DIM,
            bx + (size_t)l * H_DIM,
            xbufs[l & 1], posbufs[l & 1]);
        xc = xbufs[l & 1];
        pc = posbufs[l & 1];
    }
    final_kernel<<<(N_NODES * H_DIM + 255) / 256, 256, 0, stream>>>(x_in, xc, pc, out);
}
